// GraphSAGEModel_55783035240725
// MI455X (gfx1250) — matrix operand order repaired and checked by reading, not compiled, not run on hardware
//
#include <hip/hip_runtime.h>
#include <hip/hip_bf16.h>
#include <stdint.h>

typedef __attribute__((ext_vector_type(16))) __bf16 v16bf;
typedef __attribute__((ext_vector_type(8)))  float  v8f;

static __device__ __forceinline__ uint16_t f32_to_bf16_rne(float f) {
    union { float f; uint32_t u; } c; c.f = f;
    uint32_t u = c.u;
    uint32_t r = (u + 0x7FFFu + ((u >> 16) & 1u)) >> 16;  // round-to-nearest-even
    return (uint16_t)r;
}

// ---------------------------------------------------------------- zero fill
__global__ void fill_zero_f32(float* __restrict__ p, int n) {
    int i = (blockIdx.x * blockDim.x + threadIdx.x) * 4;
    if (i + 3 < n) {
        *(float4*)(p + i) = make_float4(0.f, 0.f, 0.f, 0.f);
    } else {
        for (int k = i; k < n; ++k) p[k] = 0.f;
    }
}

// ---------------------------------------------------------------- in-degree
__global__ void degree_kernel(const int* __restrict__ dst, int E,
                              float* __restrict__ deg, int N) {
    int e = blockIdx.x * blockDim.x + threadIdx.x;
    if (e < E) {
        unsigned d = (unsigned)dst[e];
        if (d < (unsigned)N) atomicAdd(&deg[d], 1.0f);
    }
}

// --------------------------------------------- edge scatter: agg[dst] += X[src]
// one wave32 per edge; lane handles 4 contiguous features (128 total)
__global__ void scatter_kernel(const int* __restrict__ src, const int* __restrict__ dst,
                               int E, const float* __restrict__ X,
                               float* __restrict__ agg, int N) {
    int wave = blockIdx.x * (blockDim.x >> 5) + (threadIdx.x >> 5);
    int lane = threadIdx.x & 31;
    if (wave >= E) return;
    int s = src[wave], d = dst[wave];
    if ((unsigned)s >= (unsigned)N || (unsigned)d >= (unsigned)N) return;
    float4 v = ((const float4*)(X + (size_t)s * 128))[lane];
    float* ar = agg + (size_t)d * 128 + lane * 4;
    atomicAdd(ar + 0, v.x);
    atomicAdd(ar + 1, v.y);
    atomicAdd(ar + 2, v.z);
    atomicAdd(ar + 3, v.w);
}

// --------------------------- build bf16 concat activation: [agg/deg | X]  (N x 256)
__global__ void prep_acat(const float* __restrict__ agg, const float* __restrict__ deg,
                          const float* __restrict__ X, uint16_t* __restrict__ acat, int N) {
    int i = blockIdx.x * blockDim.x + threadIdx.x;  // over N*128
    if (i >= N * 128) return;
    int nd = i >> 7, f = i & 127;
    float inv = 1.0f / fmaxf(deg[nd], 1.0f);
    size_t base = (size_t)nd * 256;
    acat[base + f]       = f32_to_bf16_rne(agg[i] * inv);
    acat[base + 128 + f] = f32_to_bf16_rne(X[i]);
}

// ------------- pre-swizzle [W_l ; W_r] (256x128) into B-fragment order (bf16)
// GEMM reads lane chunk at idx = ((colTile*8 + kTile)*32 + lane)*16 + i
// B layout (16-bit 32x16): lanes 0-15 -> n=lane, K=kt*32+{0..7,16..23} ; lanes 16-31 -> n=lane-16, K=kt*32+{8..15,24..31}
__global__ void swizzle_weights(const float* __restrict__ Wl, const float* __restrict__ Wr,
                                uint16_t* __restrict__ bswz) {
    int idx = blockIdx.x * blockDim.x + threadIdx.x;  // 8*8*32*16 = 32768
    if (idx >= 8 * 8 * 32 * 16) return;
    int i    = idx & 15;
    int lane = (idx >> 4) & 31;
    int kt   = (idx >> 9) & 7;
    int c    = idx >> 12;
    int n = c * 16 + (lane & 15);
    int k = kt * 32 + ((lane & 16) ? 8 : 0) + (i < 8 ? i : i + 8);
    float v = (k < 128) ? Wl[k * 128 + n] : Wr[(k - 128) * 128 + n];
    bswz[idx] = f32_to_bf16_rne(v);
}

// --------------------------------- fused GEMM: out = relu(Acat @ Bswz + bias)
// block = 8 waves; block -> 16-row tile, wave -> one 16-col tile; K = 256 in 8 steps of 32
__global__ void __launch_bounds__(256)
gemm_wmma_bias_relu(const uint16_t* __restrict__ acat, const uint16_t* __restrict__ bswz,
                    const float* __restrict__ bias, float* __restrict__ out, int N) {
    int lane = threadIdx.x & 31;
    int wv   = threadIdx.x >> 5;     // col tile 0..7
    int row0 = blockIdx.x * 16;

    // A-fragment layout (16-bit A 16x32): lane = row (m = lane&15);
    // lanes 0-15: K = kb..kb+7 then kb+16..kb+23 with khalf=0; lanes 16-31: khalf=8
    int m     = lane & 15;
    int khalf = (lane & 16) ? 8 : 0;
    int arow_idx = row0 + m;
    if (arow_idx >= N) arow_idx = N - 1;          // safe clamp (N is multiple of 16 anyway)
    const uint16_t* arow = acat + (size_t)arow_idx * 256;

    v8f acc = {};
#pragma unroll
    for (int kt = 0; kt < 8; ++kt) {
        int kb = kt * 32 + khalf;
        union { uint4 u[2]; v16bf v; } A;
        A.u[0] = *(const uint4*)(arow + kb);        // K kb..kb+7   (16B)
        A.u[1] = *(const uint4*)(arow + kb + 16);   // K kb+16..kb+23
        union { uint4 u[2]; v16bf v; } B;
        const uint16_t* bp = bswz + (((size_t)(wv * 8 + kt) * 32 + lane) << 4);
        B.u[0] = *(const uint4*)(bp);
        B.u[1] = *(const uint4*)(bp + 8);
        acc = __builtin_amdgcn_wmma_f32_16x16x32_bf16(
            /*neg_a=*/false, A.v, /*neg_b=*/false, B.v,
            /*c_mod=*/(short)0, acc, /*reuse_a=*/false, /*reuse_b=*/false);
    }

    // C/D layout: lane -> col n = lane&15 (+wave tile), VGPR r -> row m = r + 8*(lane>=16)
    int n   = lane & 15;
    int mb  = (lane & 16) ? 8 : 0;
    int col = wv * 16 + n;
    float bv = bias[col];
#pragma unroll
    for (int r = 0; r < 8; ++r) {
        int row = row0 + mb + r;
        if (row < N) {
            float v = acc[r] + bv;
            out[(size_t)row * 128 + col] = v > 0.f ? v : 0.f;
        }
    }
}

// ---------------------------------------------------------------- final FC (128 -> 2)
__global__ void fc_kernel(const float* __restrict__ h, const float* __restrict__ Wfc,
                          const float* __restrict__ bfc, float* __restrict__ out, int N) {
    int nd = blockIdx.x * blockDim.x + threadIdx.x;
    if (nd >= N) return;
    const float4* hr = (const float4*)(h + (size_t)nd * 128);
    float a0 = bfc[0], a1 = bfc[1];
#pragma unroll 8
    for (int i = 0; i < 32; ++i) {
        float4 v = hr[i];
        int k = i * 4;
        a0 += v.x * Wfc[(k + 0) * 2 + 0] + v.y * Wfc[(k + 1) * 2 + 0]
            + v.z * Wfc[(k + 2) * 2 + 0] + v.w * Wfc[(k + 3) * 2 + 0];
        a1 += v.x * Wfc[(k + 0) * 2 + 1] + v.y * Wfc[(k + 1) * 2 + 1]
            + v.z * Wfc[(k + 2) * 2 + 1] + v.w * Wfc[(k + 3) * 2 + 1];
    }
    out[(size_t)nd * 2 + 0] = a0;
    out[(size_t)nd * 2 + 1] = a1;
}

extern "C" void kernel_launch(void* const* d_in, const int* in_sizes, int n_in,
                              void* d_out, int out_size, void* d_ws, size_t ws_size,
                              hipStream_t stream) {
    const float* x    = (const float*)d_in[0];
    const int*   ei   = (const int*)d_in[1];
    const float* W1l  = (const float*)d_in[2];
    const float* b1   = (const float*)d_in[3];
    const float* W1r  = (const float*)d_in[4];
    const float* W2l  = (const float*)d_in[5];
    const float* b2   = (const float*)d_in[6];
    const float* W2r  = (const float*)d_in[7];
    const float* Wfc  = (const float*)d_in[8];
    const float* bfc  = (const float*)d_in[9];
    float* outp = (float*)d_out;

    const int N = in_sizes[0] / 128;
    const int E = in_sizes[1] / 2;
    const int* src = ei;
    const int* dst = ei + E;

    // ---- carve workspace (256B-aligned slices)
    char* w = (char*)d_ws;
    auto carve = [&](size_t bytes) -> void* {
        void* p = (void*)w;
        w += (bytes + 255) & ~(size_t)255;
        return p;
    };
    float*    deg   = (float*)carve((size_t)N * 4);
    float*    agg   = (float*)carve((size_t)N * 128 * 4);
    uint16_t* acat  = (uint16_t*)carve((size_t)N * 256 * 2);
    float*    h     = (float*)carve((size_t)N * 128 * 4);
    uint16_t* bswz1 = (uint16_t*)carve(8 * 8 * 32 * 16 * 2);
    uint16_t* bswz2 = (uint16_t*)carve(8 * 8 * 32 * 16 * 2);

    const int NF       = N * 128;
    const int zeroBlk  = 256;
    const int degGrid  = (E + 255) / 256;
    const int scatGrid = (E + 7) / 8;           // 8 waves per 256-thread block
    const int prepGrid = (NF + 255) / 256;
    const int gemmGrid = (N + 15) / 16;
    const int fcGrid   = (N + 255) / 256;

    // ---- degree (shared by both layers)
    fill_zero_f32<<<(N + zeroBlk * 4 - 1) / (zeroBlk * 4), zeroBlk, 0, stream>>>(deg, N);
    degree_kernel<<<degGrid, 256, 0, stream>>>(dst, E, deg, N);

    // ---- pre-swizzle both weight pairs (tiny)
    swizzle_weights<<<128, 256, 0, stream>>>(W1l, W1r, bswz1);
    swizzle_weights<<<128, 256, 0, stream>>>(W2l, W2r, bswz2);

    // ---- layer 1
    fill_zero_f32<<<(NF + zeroBlk * 4 - 1) / (zeroBlk * 4), zeroBlk, 0, stream>>>(agg, NF);
    scatter_kernel<<<scatGrid, 256, 0, stream>>>(src, dst, E, x, agg, N);
    prep_acat<<<prepGrid, 256, 0, stream>>>(agg, deg, x, acat, N);
    gemm_wmma_bias_relu<<<gemmGrid, 256, 0, stream>>>(acat, bswz1, b1, h, N);

    // ---- layer 2 (reuses agg / acat; h overwritten in place by GEMM2 — it only reads acat)
    fill_zero_f32<<<(NF + zeroBlk * 4 - 1) / (zeroBlk * 4), zeroBlk, 0, stream>>>(agg, NF);
    scatter_kernel<<<scatGrid, 256, 0, stream>>>(src, dst, E, h, agg, N);
    prep_acat<<<prepGrid, 256, 0, stream>>>(agg, deg, h, acat, N);
    gemm_wmma_bias_relu<<<gemmGrid, 256, 0, stream>>>(acat, bswz2, b2, h, N);

    // ---- head
    fc_kernel<<<fcGrid, 256, 0, stream>>>(h, Wfc, bfc, outp, N);
}
